// CausalSelfAttention_15564961480737
// MI455X (gfx1250) — matrix operand order repaired and checked by reading, not compiled, not run on hardware
//
#include <hip/hip_runtime.h>
#include <hip/hip_bf16.h>

typedef __attribute__((ext_vector_type(16))) _Float16 v16h;
typedef __attribute__((ext_vector_type(8)))  _Float16 v8h;
typedef __attribute__((ext_vector_type(8)))  float    v8f;

union av16 { v16h v; v8h h[2]; };

#define B_BATCH 2
#define T_SEQ   2048
#define C_DIM   1024
#define H_HEADS 16
#define D_HEAD  64

// ---------------------------------------------------------------------------
// f32 -> f16 conversion
// ---------------------------------------------------------------------------
__global__ __launch_bounds__(256) void cvt_f32_f16(const float* __restrict__ in,
                                                   _Float16* __restrict__ out, int n) {
    int i = blockIdx.x * 256 + threadIdx.x;
    if (i < n) out[i] = (_Float16)in[i];
}

// ---------------------------------------------------------------------------
// RoPE (in place) on a [BH, T, D] f16 buffer. One thread per (bh,t,d<32) pair.
// emb = concat(freqs,freqs) => cos/sin identical for d and d+32.
// ---------------------------------------------------------------------------
__global__ __launch_bounds__(256) void rope_f16(_Float16* __restrict__ q, int total) {
    int idx = blockIdx.x * 256 + threadIdx.x;
    if (idx >= total) return;
    const int d  = idx & 31;
    const int t  = (idx >> 5) & (T_SEQ - 1);
    const int bh = idx >> 16;                       // 32*2048 = 65536
    _Float16* base = q + ((size_t)bh * T_SEQ + t) * D_HEAD;
    // inv_freq = 10000^(-d/32) ; ln(10000) = 9.210340371976184
    const float inv = __expf(-(float)d * (9.210340371976184f / 32.0f));
    const float ang = (float)t * inv;
    const float s = __sinf(ang), c = __cosf(ang);
    const float x0 = (float)base[d];
    const float x1 = (float)base[d + 32];
    base[d]      = (_Float16)(x0 * c - x1 * s);
    base[d + 32] = (_Float16)(x1 * c + x0 * s);
}

// ---------------------------------------------------------------------------
// WMMA GEMM: y = A[M,K](f16) * W[N,K](f16)^T + bias[N](f32)
// One wave computes a 32x32 output block as a 2x2 grid of 16x16 WMMA tiles:
// 4 v_wmma per K-step against 2 A-fragments + 2 B-fragments (2x data reuse).
// mode 0: f16 out scattered to [B,H,T,D]      (q,k)
// mode 1: f16 out scattered to [B,H,D,T]      (v, transposed for P*V loads)
// mode 2: f32 out row-major [M,N]             (final projection -> d_out)
// ---------------------------------------------------------------------------
__global__ __launch_bounds__(256) void gemm32_wmma(
    const _Float16* __restrict__ A,
    const _Float16* __restrict__ W,
    const float*    __restrict__ bias,
    void*           __restrict__ out,
    int M, int N, int K, int mode)
{
    const int lane = threadIdx.x & 31;
    const int wave = threadIdx.x >> 5;
    const int hl   = lane >> 4;        // half-wave select
    const int l15  = lane & 15;
    const int Nt   = N >> 5;           // 32-wide tiles in N
    const int tile = blockIdx.x * 8 + wave;
    const int m0   = (tile / Nt) << 5;
    const int n0   = (tile % Nt) << 5;

    // A layout: lane = row; elem j -> K = k0 + 16*hl + j
    const _Float16* arow0 = A + (size_t)(m0 + l15)      * K + (hl << 4);
    const _Float16* arow1 = A + (size_t)(m0 + 16 + l15) * K + (hl << 4);
    // B = W^T: lane = col; elem j -> K = k0 + 16*hl + j  (contiguous in W row)
    const _Float16* brow0 = W + (size_t)(n0 + l15)      * K + (hl << 4);
    const _Float16* brow1 = W + (size_t)(n0 + 16 + l15) * K + (hl << 4);

    v8f acc00 = {}, acc01 = {}, acc10 = {}, acc11 = {};
    for (int k0 = 0; k0 < K; k0 += 32) {
        av16 a0, a1, b0, b1;
        a0.h[0] = *(const v8h*)(arow0 + k0);
        a0.h[1] = *(const v8h*)(arow0 + k0 + 8);
        a1.h[0] = *(const v8h*)(arow1 + k0);
        a1.h[1] = *(const v8h*)(arow1 + k0 + 8);
        b0.h[0] = *(const v8h*)(brow0 + k0);
        b0.h[1] = *(const v8h*)(brow0 + k0 + 8);
        b1.h[0] = *(const v8h*)(brow1 + k0);
        b1.h[1] = *(const v8h*)(brow1 + k0 + 8);
        __builtin_prefetch(arow0 + k0 + 512, 0, 1);
        __builtin_prefetch(arow1 + k0 + 512, 0, 1);
        __builtin_prefetch(brow0 + k0 + 512, 0, 1);
        __builtin_prefetch(brow1 + k0 + 512, 0, 1);
        acc00 = __builtin_amdgcn_wmma_f32_16x16x32_f16(false, a0.v, false, b0.v, (short)0, acc00, false, false);
        acc01 = __builtin_amdgcn_wmma_f32_16x16x32_f16(false, a0.v, false, b1.v, (short)0, acc01, false, false);
        acc10 = __builtin_amdgcn_wmma_f32_16x16x32_f16(false, a1.v, false, b0.v, (short)0, acc10, false, false);
        acc11 = __builtin_amdgcn_wmma_f32_16x16x32_f16(false, a1.v, false, b1.v, (short)0, acc11, false, false);
    }

    const float bcol0 = bias[n0 + l15];
    const float bcol1 = bias[n0 + 16 + l15];
    #pragma unroll
    for (int i = 0; i < 2; ++i) {
        #pragma unroll
        for (int j = 0; j < 2; ++j) {
            const v8f& acc = (i == 0) ? (j == 0 ? acc00 : acc01)
                                      : (j == 0 ? acc10 : acc11);
            const float bcol = j ? bcol1 : bcol0;
            const int   nn   = n0 + (j << 4) + l15;
            if (mode == 2) {
                float* o = (float*)out;
                #pragma unroll
                for (int r = 0; r < 8; ++r) {
                    const int m = m0 + (i << 4) + r + (hl << 3);  // C layout: M = r + 8*hl
                    o[(size_t)m * N + nn] = acc[r] + bcol;
                }
            } else {
                _Float16* o = (_Float16*)out;
                const int hh = nn >> 6, d = nn & 63;
                #pragma unroll
                for (int r = 0; r < 8; ++r) {
                    const int m   = m0 + (i << 4) + r + (hl << 3);
                    const int bat = m >> 11, t = m & (T_SEQ - 1);
                    const float val = acc[r] + bcol;
                    const size_t idx = (mode == 0)
                        ? ((((size_t)bat * H_HEADS + hh) * T_SEQ + t) * D_HEAD + d)
                        : ((((size_t)bat * H_HEADS + hh) * D_HEAD + d) * T_SEQ + t);
                    o[idx] = (_Float16)val;
                }
            }
        }
    }
}

// ---------------------------------------------------------------------------
// Flash-style causal attention. One wave owns a 16-row Q tile, streams 32 keys
// per iteration. Scores and P*V both on v_wmma_f32_16x16x32_f16.
// q,k: [BH,T,D]   v: [BH,D,T] (pre-transposed)   y: [B,T,C] f16
// ---------------------------------------------------------------------------
__global__ __launch_bounds__(256) void attn_wmma(
    const _Float16* __restrict__ qh,
    const _Float16* __restrict__ kh,
    const _Float16* __restrict__ vt,
    _Float16*       __restrict__ yh)
{
    __shared__ __align__(32) _Float16 pbuf[8][16 * 32];   // per-wave P staging
    const int lane = threadIdx.x & 31;
    const int wave = threadIdx.x >> 5;
    const int hl   = lane >> 4;
    const int l15  = lane & 15;
    const int wg   = blockIdx.x * 8 + wave;               // 0..4095
    const int qi   = wg & (T_SEQ / 16 - 1);
    const int bh   = wg >> 7;
    const int t0   = qi << 4;
    const int bat  = bh >> 4;
    const int hh   = bh & 15;

    const _Float16* qrow  = qh + ((size_t)bh * T_SEQ + t0 + l15) * D_HEAD + (hl << 4);
    const _Float16* kbase = kh + (size_t)bh * T_SEQ * D_HEAD;
    const _Float16* vbase = vt + (size_t)bh * D_HEAD * T_SEQ;

    // Q fragments for d=0..31 and d=32..63 (reused across the whole K loop)
    av16 qa0, qa1;
    qa0.h[0] = *(const v8h*)(qrow +  0);
    qa0.h[1] = *(const v8h*)(qrow +  8);
    qa1.h[0] = *(const v8h*)(qrow + 32);
    qa1.h[1] = *(const v8h*)(qrow + 40);

    float mrow[8], lrow[8];
    v8f o[4];
    #pragma unroll
    for (int r = 0; r < 8; ++r) {
        mrow[r] = -1e30f; lrow[r] = 0.f;
        o[0][r] = 0.f; o[1][r] = 0.f; o[2][r] = 0.f; o[3][r] = 0.f;
    }

    for (int s0 = 0; s0 < t0 + 16; s0 += 32) {
        v8f sc0, sc1;
        {   // score subtile 0: keys s0..s0+15
            const _Float16* krow = kbase + (size_t)(s0 + l15) * D_HEAD + (hl << 4);
            av16 b0, b1;
            b0.h[0] = *(const v8h*)(krow +  0);
            b0.h[1] = *(const v8h*)(krow +  8);
            b1.h[0] = *(const v8h*)(krow + 32);
            b1.h[1] = *(const v8h*)(krow + 40);
            v8f a = {};
            a = __builtin_amdgcn_wmma_f32_16x16x32_f16(false, qa0.v, false, b0.v, (short)0, a, false, false);
            a = __builtin_amdgcn_wmma_f32_16x16x32_f16(false, qa1.v, false, b1.v, (short)0, a, false, false);
            sc0 = a;
        }
        {   // score subtile 1: keys s0+16..s0+31
            const _Float16* krow = kbase + (size_t)(s0 + 16 + l15) * D_HEAD + (hl << 4);
            av16 b0, b1;
            b0.h[0] = *(const v8h*)(krow +  0);
            b0.h[1] = *(const v8h*)(krow +  8);
            b1.h[0] = *(const v8h*)(krow + 32);
            b1.h[1] = *(const v8h*)(krow + 40);
            v8f a = {};
            a = __builtin_amdgcn_wmma_f32_16x16x32_f16(false, qa0.v, false, b0.v, (short)0, a, false, false);
            a = __builtin_amdgcn_wmma_f32_16x16x32_f16(false, qa1.v, false, b1.v, (short)0, a, false, false);
            sc1 = a;
        }
        // scale 1/sqrt(D) + causal mask (element (r,lane): row t0+r+8*hl, col s0+{0,16}+l15)
        #pragma unroll
        for (int r = 0; r < 8; ++r) {
            const int t = t0 + r + (hl << 3);
            sc0[r] = (s0 + l15      <= t) ? sc0[r] * 0.125f : -1e30f;
            sc1[r] = (s0 + 16 + l15 <= t) ? sc1[r] * 0.125f : -1e30f;
        }
        // online softmax: rows live in 16-lane half-groups -> xor 1/2/4/8 reduce
        float corr[8];
        #pragma unroll
        for (int r = 0; r < 8; ++r) {
            float v = fmaxf(sc0[r], sc1[r]);
            v = fmaxf(v, __shfl_xor(v, 1, 32));
            v = fmaxf(v, __shfl_xor(v, 2, 32));
            v = fmaxf(v, __shfl_xor(v, 4, 32));
            v = fmaxf(v, __shfl_xor(v, 8, 32));
            const float nm = fmaxf(mrow[r], v);
            corr[r] = __expf(mrow[r] - nm);
            mrow[r] = nm;
        }
        #pragma unroll
        for (int r = 0; r < 8; ++r) {
            const float p0 = __expf(sc0[r] - mrow[r]);
            const float p1 = __expf(sc1[r] - mrow[r]);
            sc0[r] = p0; sc1[r] = p1;
            float v = p0 + p1;
            v += __shfl_xor(v, 1, 32);
            v += __shfl_xor(v, 2, 32);
            v += __shfl_xor(v, 4, 32);
            v += __shfl_xor(v, 8, 32);
            lrow[r] = lrow[r] * corr[r] + v;
        }
        #pragma unroll
        for (int r = 0; r < 8; ++r) {
            o[0][r] *= corr[r]; o[1][r] *= corr[r];
            o[2][r] *= corr[r]; o[3][r] *= corr[r];
        }
        // C-layout -> A-layout conversion of P via LDS
        _Float16* pb = pbuf[wave];
        #pragma unroll
        for (int r = 0; r < 8; ++r) {
            const int row = r + (hl << 3);
            pb[row * 32 + l15]      = (_Float16)sc0[r];
            pb[row * 32 + 16 + l15] = (_Float16)sc1[r];
        }
        asm volatile("s_wait_dscnt 0" ::: "memory");
        av16 pa;
        {
            const _Float16* prow = pb + l15 * 32 + (hl << 4);
            pa.h[0] = *(const v8h*)(prow);
            pa.h[1] = *(const v8h*)(prow + 8);
        }
        // O += P (16x32) @ V (32x64): V pre-transposed [D,T] -> contiguous B loads
        #pragma unroll
        for (int j = 0; j < 4; ++j) {
            const _Float16* vrow = vbase + (size_t)(j * 16 + l15) * T_SEQ + s0 + (hl << 4);
            av16 vb;
            vb.h[0] = *(const v8h*)(vrow);
            vb.h[1] = *(const v8h*)(vrow + 8);
            o[j] = __builtin_amdgcn_wmma_f32_16x16x32_f16(false, pa.v, false, vb.v, (short)0, o[j], false, false);
        }
    }

    // normalize and write y[B,T,C] (c = hh*64 + j*16 + l15)
    _Float16* ybase = yh + ((size_t)bat * T_SEQ) * C_DIM + hh * D_HEAD;
    #pragma unroll
    for (int r = 0; r < 8; ++r) {
        const int t = t0 + r + (hl << 3);
        const float inv_l = 1.0f / lrow[r];
        ybase[(size_t)t * C_DIM +  0 + l15] = (_Float16)(o[0][r] * inv_l);
        ybase[(size_t)t * C_DIM + 16 + l15] = (_Float16)(o[1][r] * inv_l);
        ybase[(size_t)t * C_DIM + 32 + l15] = (_Float16)(o[2][r] * inv_l);
        ybase[(size_t)t * C_DIM + 48 + l15] = (_Float16)(o[3][r] * inv_l);
    }
}

// ---------------------------------------------------------------------------
extern "C" void kernel_launch(void* const* d_in, const int* in_sizes, int n_in,
                              void* d_out, int out_size, void* d_ws, size_t ws_size,
                              hipStream_t stream) {
    const float* x  = (const float*)d_in[0];
    const float* Wq = (const float*)d_in[1];
    const float* bq = (const float*)d_in[2];
    const float* Wk = (const float*)d_in[3];
    const float* bk = (const float*)d_in[4];
    const float* Wv = (const float*)d_in[5];
    const float* bv = (const float*)d_in[6];
    const float* Wp = (const float*)d_in[7];
    const float* bp = (const float*)d_in[8];

    const int M  = B_BATCH * T_SEQ;          // 4096
    const size_t nX = (size_t)M * C_DIM;     // 4,194,304
    const size_t nW = (size_t)C_DIM * C_DIM; // 1,048,576

    char* ws = (char*)d_ws;
    size_t off = 0;
    auto alloc_h = [&](size_t elems) {
        _Float16* p = (_Float16*)(ws + off);
        off = (off + elems * sizeof(_Float16) + 255) & ~(size_t)255;
        return p;
    };
    _Float16* xh  = alloc_h(nX);
    _Float16* wqh = alloc_h(nW);
    _Float16* wkh = alloc_h(nW);
    _Float16* wvh = alloc_h(nW);
    _Float16* wph = alloc_h(nW);
    _Float16* qh  = alloc_h(nX);   // [B,H,T,D]
    _Float16* kh  = alloc_h(nX);   // [B,H,T,D]
    _Float16* vth = alloc_h(nX);   // [B,H,D,T]
    _Float16* yh  = alloc_h(nX);   // [B,T,C]

    cvt_f32_f16<<<(int)((nX + 255) / 256), 256, 0, stream>>>(x,  xh,  (int)nX);
    cvt_f32_f16<<<(int)((nW + 255) / 256), 256, 0, stream>>>(Wq, wqh, (int)nW);
    cvt_f32_f16<<<(int)((nW + 255) / 256), 256, 0, stream>>>(Wk, wkh, (int)nW);
    cvt_f32_f16<<<(int)((nW + 255) / 256), 256, 0, stream>>>(Wv, wvh, (int)nW);
    cvt_f32_f16<<<(int)((nW + 255) / 256), 256, 0, stream>>>(Wp, wph, (int)nW);

    const int gemmBlocks = (M / 32) * (C_DIM / 32) / 8;   // 512, exact
    gemm32_wmma<<<gemmBlocks, 256, 0, stream>>>(xh, wqh, bq, qh,  M, C_DIM, C_DIM, 0);
    gemm32_wmma<<<gemmBlocks, 256, 0, stream>>>(xh, wkh, bk, kh,  M, C_DIM, C_DIM, 0);
    gemm32_wmma<<<gemmBlocks, 256, 0, stream>>>(xh, wvh, bv, vth, M, C_DIM, C_DIM, 1);

    const int nrope = B_BATCH * H_HEADS * T_SEQ * (D_HEAD / 2); // 2,097,152
    rope_f16<<<nrope / 256, 256, 0, stream>>>(qh, nrope);
    rope_f16<<<nrope / 256, 256, 0, stream>>>(kh, nrope);

    const int attnBlocks = (B_BATCH * H_HEADS * (T_SEQ / 16)) / 8; // 512, exact
    attn_wmma<<<attnBlocks, 256, 0, stream>>>(qh, kh, vth, yh);

    gemm32_wmma<<<gemmBlocks, 256, 0, stream>>>(yh, wph, bp, d_out, M, C_DIM, C_DIM, 2);
}
